// DynamicReductionNetworkJit_34505767256549
// MI455X (gfx1250) — matrix operand order repaired and checked by reading, not compiled, not run on hardware
//
#include <hip/hip_runtime.h>
#include <hip/hip_bf16.h>

typedef __attribute__((ext_vector_type(16))) _Float16 v16h;
typedef __attribute__((ext_vector_type(8)))  float    v8f;

#define HDIM   64
#define KNN    16
#define NGRAPH 16
#define NEGBIG (-1e30f)

#if defined(__AMDGCN__)
#define HAS_ASYNC_LDS 1
#endif

// ---------------------------------------------------------------------------
// Graph offsets: gstart[b] = first node of graph b (batch sorted), gstart[B]=n
// ---------------------------------------------------------------------------
__global__ void k_graph_offsets(const int* __restrict__ batch, int n, int* __restrict__ gstart) {
    int t = threadIdx.x;
    if (t <= NGRAPH) gstart[t] = n;
    __syncthreads();
    for (int i = t; i < n; i += blockDim.x) atomicMin(&gstart[batch[i]], i);
    __threadfence();
    __syncthreads();
    if (t == 0) {
        for (int b = NGRAPH - 1; b >= 0; --b)
            if (gstart[b] > gstart[b + 1]) gstart[b] = gstart[b + 1];
    }
}

__global__ void k_fill_i32(int* __restrict__ p, int n, int v) {
    int i = blockIdx.x * blockDim.x + threadIdx.x;
    if (i < n) p[i] = v;
}

__global__ void k_f2h(const float* __restrict__ s, _Float16* __restrict__ d, int n) {
    int i = blockIdx.x * blockDim.x + threadIdx.x;
    if (i < n) d[i] = (_Float16)s[i];
}

// ---------------------------------------------------------------------------
// inputnet: h = relu((x*datanorm) @ in_w.T + in_b)   [n,4] -> [n,64] (f32+f16)
// ---------------------------------------------------------------------------
__global__ void k_inputnet(const float* __restrict__ x, const float* __restrict__ datanorm,
                           const float* __restrict__ in_w, const float* __restrict__ in_b,
                           float* __restrict__ h, _Float16* __restrict__ hh, int n) {
    int t = blockIdx.x * blockDim.x + threadIdx.x;
    if (t >= n * HDIM) return;
    int i = t >> 6, c = t & 63;
    float acc = in_b[c];
#pragma unroll
    for (int k = 0; k < 4; ++k) acc += x[i * 4 + k] * datanorm[k] * in_w[c * 4 + k];
    acc = fmaxf(acc, 0.0f);
    h[t] = acc;
    hh[t] = (_Float16)acc;
}

__global__ void k_sqnorm(const float* __restrict__ h, float* __restrict__ sq, int n) {
    int i = blockIdx.x * blockDim.x + threadIdx.x;
    if (i >= n) return;
    float s = 0.0f;
#pragma unroll 8
    for (int c = 0; c < HDIM; ++c) { float v = h[i * HDIM + c]; s += v * v; }
    sq[i] = s;
}

// ---------------------------------------------------------------------------
// kNN: one wave per 16-query tile. Gram tiles via v_wmma_f32_16x16x32_f16,
// candidate tiles streamed memory->LDS with GLOBAL_LOAD_ASYNC_TO_LDS_B128
// (ASYNCcnt), dist = |qi|^2 + |cj|^2 - 2*dot masked per-graph + validity,
// top-16 kept per row in registers (sorted ascending, matching top_k(-d)).
// ---------------------------------------------------------------------------
__global__ void k_knn(const _Float16* __restrict__ hh, const float* __restrict__ sq,
                      const int* __restrict__ batch, const int* __restrict__ valid,
                      const int* __restrict__ gstart, int* __restrict__ idx) {
    __shared__ alignas(32) _Float16 qt[16 * 64];
    __shared__ alignas(32) _Float16 ct[16 * 64];
    __shared__ float sqq[16], sqc[16], dist[16][16];
    __shared__ int   qb[16], cb[16], cj[16];

    const int lane = threadIdx.x;            // 0..31 (wave32)
    const int q0 = blockIdx.x * 16;

    for (int t = lane; t < 16 * 64; t += 32) qt[t] = hh[q0 * 64 + t];
    if (lane < 16) { sqq[lane] = sq[q0 + lane]; qb[lane] = batch[q0 + lane]; }
    __syncthreads();

    const int cs = gstart[qb[0]];
    const int ce = gstart[qb[15] + 1];

    float bd[KNN]; int bi[KNN];
#pragma unroll
    for (int s = 0; s < KNN; ++s) { bd[s] = __builtin_inff(); bi[s] = q0 + (lane < 16 ? lane : 0); }

    // A slices are invariant over candidate tiles
    const int aoff = (lane & 15) * 64 + (lane >> 4) * 16;
    v16h a0 = *(const v16h*)&qt[aoff];
    v16h a1 = *(const v16h*)&qt[aoff + 32];

#ifdef HAS_ASYNC_LDS
    const unsigned int ct_lds = (unsigned int)(size_t)&ct[0] + (unsigned int)lane * 16u;
#endif

    for (int c0 = cs; c0 < ce; c0 += 16) {
        __builtin_prefetch(hh + (size_t)(c0 + 16) * HDIM, 0, 1);  // global_prefetch_b8
        __syncthreads();   // prior ds reads of ct/dist complete before overwrite

#ifdef HAS_ASYNC_LDS
        {
            // 16x64 f16 tile = 2KB: 4 x (32 lanes x 16B) async copies, INST_OFFSET
            // advances both global and LDS addresses (ISA 15.18.3).
            unsigned long long g =
                (unsigned long long)(size_t)(hh + (size_t)c0 * HDIM) + (unsigned long long)lane * 16u;
            asm volatile(
                "global_load_async_to_lds_b128 %0, %1, off\n\t"
                "global_load_async_to_lds_b128 %0, %1, off offset:512\n\t"
                "global_load_async_to_lds_b128 %0, %1, off offset:1024\n\t"
                "global_load_async_to_lds_b128 %0, %1, off offset:1536"
                :: "v"(ct_lds), "v"(g) : "memory");
        }
#else
        for (int t = lane; t < 16 * 64; t += 32) {
            int j = c0 + (t >> 6);
            ct[t] = (j < ce) ? hh[j * 64 + (t & 63)] : (_Float16)0.0f;
        }
#endif
        if (lane < 16) {
            int j = c0 + lane;
            bool in = j < ce;
            sqc[lane] = in ? sq[j] : 0.0f;
            cb[lane]  = (in && valid[j]) ? batch[j] : -1;   // tail/invalid -> masked below
            cj[lane]  = in ? j : q0;
        }
#ifdef HAS_ASYNC_LDS
        asm volatile("s_wait_asynccnt 0" ::: "memory");
#endif
        __syncthreads();

        v8f acc = {};
        v16h b0 = *(const v16h*)&ct[aoff];
        v16h b1 = *(const v16h*)&ct[aoff + 32];
        acc = __builtin_amdgcn_wmma_f32_16x16x32_f16(false, a0, false, b0, (short)0, acc, false, false);
        acc = __builtin_amdgcn_wmma_f32_16x16x32_f16(false, a1, false, b1, (short)0, acc, false, false);

        const int ncol = lane & 15;
        const int mofs = (lane & 16) ? 8 : 0;
#pragma unroll
        for (int r = 0; r < 8; ++r) {
            int m = r + mofs;
            float d = sqq[m] + sqc[ncol] - 2.0f * acc[r];
            if (cb[ncol] != qb[m]) d = __builtin_inff();   // cross-graph / invalid / tail
            dist[m][ncol] = d;
        }
        __syncthreads();

        if (lane < 16) {
#pragma unroll
            for (int t = 0; t < 16; ++t) {
                float d = dist[lane][t];
                int   j = cj[t];
                if (d < bd[KNN - 1]) {
                    bd[KNN - 1] = d; bi[KNN - 1] = j;
#pragma unroll
                    for (int s = KNN - 1; s > 0; --s) {
                        if (bd[s] < bd[s - 1]) {
                            float td = bd[s]; bd[s] = bd[s - 1]; bd[s - 1] = td;
                            int   ti = bi[s]; bi[s] = bi[s - 1]; bi[s - 1] = ti;
                        }
                    }
                }
            }
        }
    }
    if (lane < 16) {
#pragma unroll
        for (int s = 0; s < KNN; ++s) idx[(q0 + lane) * KNN + s] = bi[s];
    }
}

// ---------------------------------------------------------------------------
// Fused EdgeConv: one wave per node. Edge tile E[16,128] = [x_i, x_j - x_i]
// staged in LDS -> GEMM1 (128x128, WMMA) -> BN+ReLU -> GEMM2 (128x64, WMMA)
// -> BN+ReLU -> sum over K=16 edges (column reduce) -> h_out[node,64].
// ---------------------------------------------------------------------------
__global__ void k_edgeconv(const float* __restrict__ hin, const int* __restrict__ idx,
                           const int* __restrict__ valid,
                           const _Float16* __restrict__ w0h, const float* __restrict__ l0_b,
                           const float* __restrict__ bn0_g, const float* __restrict__ bn0_b,
                           const _Float16* __restrict__ w1h, const float* __restrict__ l1_b,
                           const float* __restrict__ bn1_g, const float* __restrict__ bn1_b,
                           float* __restrict__ hout) {
    __shared__ alignas(32) _Float16 E [16 * 128];
    __shared__ alignas(32) _Float16 Mi[16 * 128];
    __shared__ float red[32];

    const int lane = threadIdx.x;
    const int nd = blockIdx.x;
    const float rs = rsqrtf(1.0f + 1e-5f);     // eval-mode BN scale base

    // Build edge tile: even lanes -> x_i half, odd lanes -> (x_j - x_i) half
    {
        int r = lane >> 1;
        int j = idx[nd * KNN + r];
        bool hi = lane & 1;
        int base = r * 128 + (hi ? 64 : 0);
#pragma unroll 8
        for (int e = 0; e < 64; ++e) {
            float xi = hin[nd * 64 + e];
            float v = hi ? (hin[(size_t)j * 64 + e] - xi) : xi;
            E[base + e] = (_Float16)v;
        }
    }
    __syncthreads();

    const int arow = (lane & 15);
    const int acol = (lane >> 4) * 16;
    const int ncol = lane & 15;
    const int mofs = (lane & 16) ? 8 : 0;

    // ---- GEMM1: [16,128] x [128,128]^T -> Mi[16,128] (BN0 + ReLU) ----
    v16h ea[4];
#pragma unroll
    for (int kt = 0; kt < 4; ++kt) ea[kt] = *(const v16h*)&E[arow * 128 + kt * 32 + acol];

    for (int ct = 0; ct < 8; ++ct) {
        v8f acc = {};
#pragma unroll
        for (int kt = 0; kt < 4; ++kt) {
            v16h b = *(const v16h*)&w0h[(size_t)(ct * 16 + arow) * 128 + kt * 32 + acol];
            acc = __builtin_amdgcn_wmma_f32_16x16x32_f16(false, ea[kt], false, b, (short)0, acc, false, false);
        }
        int col = ct * 16 + ncol;
        float sc = bn0_g[col] * rs, sb = bn0_b[col], lb = l0_b[col];
#pragma unroll
        for (int r = 0; r < 8; ++r) {
            float v = fmaxf((acc[r] + lb) * sc + sb, 0.0f);
            Mi[(r + mofs) * 128 + col] = (_Float16)v;
        }
    }
    __syncthreads();

    // ---- GEMM2: [16,128] x [128,64]^T -> BN1 + ReLU -> sum over 16 edges ----
    v16h ma[4];
#pragma unroll
    for (int kt = 0; kt < 4; ++kt) ma[kt] = *(const v16h*)&Mi[arow * 128 + kt * 32 + acol];

    const bool nvalid = valid[nd] != 0;
    for (int ct = 0; ct < 4; ++ct) {
        v8f acc = {};
#pragma unroll
        for (int kt = 0; kt < 4; ++kt) {
            v16h b = *(const v16h*)&w1h[(size_t)(ct * 16 + arow) * 128 + kt * 32 + acol];
            acc = __builtin_amdgcn_wmma_f32_16x16x32_f16(false, ma[kt], false, b, (short)0, acc, false, false);
        }
        int col = ct * 16 + ncol;
        float sc = bn1_g[col] * rs, sb = bn1_b[col], lb = l1_b[col];
        float part = 0.0f;
#pragma unroll
        for (int r = 0; r < 8; ++r) part += fmaxf((acc[r] + lb) * sc + sb, 0.0f);
        red[((lane & 16) ? 16 : 0) + ncol] = part;
        __syncthreads();
        if (lane < 16) {
            float v = nvalid ? (red[lane] + red[16 + lane]) : 0.0f;
            hout[(size_t)nd * 64 + ct * 16 + lane] = v;
        }
        __syncthreads();
    }
}

// ---------------------------------------------------------------------------
// Graclus matching: partner = idx[argmax_k w], w = (j==i ? -inf : ||hj-hi||*2/K)
// ---------------------------------------------------------------------------
__global__ void k_partner(const float* __restrict__ h, const int* __restrict__ idx,
                          const int* __restrict__ valid, int* __restrict__ partner, int n) {
    int i = blockIdx.x * blockDim.x + threadIdx.x;
    if (i >= n) return;
    float best = -__builtin_inff();
    int barg = 0;
#pragma unroll
    for (int k = 0; k < KNN; ++k) {
        int j = idx[i * KNN + k];
        float s = 1e-12f;
#pragma unroll 8
        for (int c = 0; c < HDIM; ++c) {
            float d = h[(size_t)j * HDIM + c] - h[(size_t)i * HDIM + c];
            s += d * d;
        }
        float w = (j == i) ? -__builtin_inff() : sqrtf(s) * (2.0f / (float)KNN);
        if (w > best) { best = w; barg = k; }   // first max (jnp.argmax)
    }
    int p = idx[i * KNN + barg];
    if (!(valid[i] && valid[p])) p = i;
    partner[i] = p;
}

__global__ void k_pool(const float* __restrict__ h, const int* __restrict__ partner,
                       const int* __restrict__ valid, float* __restrict__ hout,
                       _Float16* __restrict__ hhout, int* __restrict__ validout, int n) {
    int t = blockIdx.x * blockDim.x + threadIdx.x;
    if (t >= n * HDIM) return;
    int i = t >> 6, c = t & 63;
    int p = partner[i];
    bool mutual = (partner[p] == i);
    int cluster = mutual ? (i < p ? i : p) : i;
    bool v = valid[i] && (cluster == i);
    float out = 0.0f;
    if (v) {
        out = h[t];
        if (mutual && p != i) out = fmaxf(out, h[(size_t)p * HDIM + c]);
    }
    hout[t] = out;
    hhout[t] = (_Float16)out;
    if (c == 0) validout[i] = v ? 1 : 0;
}

// ---------------------------------------------------------------------------
// Global max pool per graph + 64->64->64->1 MLP head. One block, 16x64 threads.
// ---------------------------------------------------------------------------
__global__ void k_output(const float* __restrict__ h, const int* __restrict__ valid,
                         const int* __restrict__ gstart,
                         const float* __restrict__ o0_w, const float* __restrict__ o0_b,
                         const float* __restrict__ o1_w, const float* __restrict__ o1_b,
                         const float* __restrict__ o2_w, const float* __restrict__ o2_b,
                         float* __restrict__ out) {
    __shared__ float G[NGRAPH][HDIM], G1[NGRAPH][HDIM], G2[NGRAPH][HDIM];
    int t = threadIdx.x;
    int b = t >> 6, c = t & 63;
    float m = NEGBIG;
    for (int i = gstart[b]; i < gstart[b + 1]; ++i)
        if (valid[i]) m = fmaxf(m, h[(size_t)i * HDIM + c]);
    G[b][c] = m;
    __syncthreads();
    float a = o0_b[c];
#pragma unroll 8
    for (int k = 0; k < HDIM; ++k) a += G[b][k] * o0_w[c * HDIM + k];
    G1[b][c] = fmaxf(a, 0.0f);
    __syncthreads();
    a = o1_b[c];
#pragma unroll 8
    for (int k = 0; k < HDIM; ++k) a += G1[b][k] * o1_w[c * HDIM + k];
    G2[b][c] = fmaxf(a, 0.0f);
    __syncthreads();
    if (c == 0) {
        float s = o2_b[0];
#pragma unroll 8
        for (int k = 0; k < HDIM; ++k) s += G2[b][k] * o2_w[k];
        out[b] = s;
    }
}

// ---------------------------------------------------------------------------
extern "C" void kernel_launch(void* const* d_in, const int* in_sizes, int n_in,
                              void* d_out, int out_size, void* d_ws, size_t ws_size,
                              hipStream_t stream) {
    const float* x        = (const float*)d_in[0];
    const int*   batch    = (const int*)  d_in[1];
    const float* datanorm = (const float*)d_in[2];
    const float* in_w     = (const float*)d_in[3];
    const float* in_b     = (const float*)d_in[4];
    const float* a_l0_w[2]  = { (const float*)d_in[5],  (const float*)d_in[13] };
    const float* a_l0_b[2]  = { (const float*)d_in[6],  (const float*)d_in[14] };
    const float* a_bn0_g[2] = { (const float*)d_in[7],  (const float*)d_in[15] };
    const float* a_bn0_b[2] = { (const float*)d_in[8],  (const float*)d_in[16] };
    const float* a_l1_w[2]  = { (const float*)d_in[9],  (const float*)d_in[17] };
    const float* a_l1_b[2]  = { (const float*)d_in[10], (const float*)d_in[18] };
    const float* a_bn1_g[2] = { (const float*)d_in[11], (const float*)d_in[19] };
    const float* a_bn1_b[2] = { (const float*)d_in[12], (const float*)d_in[20] };
    const float* o0_w = (const float*)d_in[21];
    const float* o0_b = (const float*)d_in[22];
    const float* o1_w = (const float*)d_in[23];
    const float* o1_b = (const float*)d_in[24];
    const float* o2_w = (const float*)d_in[25];
    const float* o2_b = (const float*)d_in[26];
    float* out = (float*)d_out;

    const int n = in_sizes[0] / 4;   // 16384

    // ---- workspace carve-up (256B-aligned) ----
    size_t off = 0;
    auto take = [&](size_t bytes) -> void* {
        void* p = (char*)d_ws + off;
        off += (bytes + 255) & ~(size_t)255;
        return p;
    };
    int*      gstart  = (int*)take((NGRAPH + 1) * sizeof(int));
    float*    hA      = (float*)take((size_t)n * HDIM * 4);
    _Float16* hAh     = (_Float16*)take((size_t)n * HDIM * 2);
    float*    hB      = (float*)take((size_t)n * HDIM * 4);
    float*    sq      = (float*)take((size_t)n * 4);
    int*      idx     = (int*)take((size_t)n * KNN * 4);
    int*      validA  = (int*)take((size_t)n * 4);
    int*      validB  = (int*)take((size_t)n * 4);
    int*      partner = (int*)take((size_t)n * 4);
    _Float16* w0h[2]  = { (_Float16*)take(128 * 128 * 2), (_Float16*)take(128 * 128 * 2) };
    _Float16* w1h[2]  = { (_Float16*)take(64 * 128 * 2),  (_Float16*)take(64 * 128 * 2) };

    // ---- setup ----
    k_graph_offsets<<<1, 256, 0, stream>>>(batch, n, gstart);
    k_fill_i32<<<(n + 255) / 256, 256, 0, stream>>>(validA, n, 1);
    for (int l = 0; l < 2; ++l) {
        k_f2h<<<(128 * 128 + 255) / 256, 256, 0, stream>>>(a_l0_w[l], w0h[l], 128 * 128);
        k_f2h<<<(64 * 128 + 255) / 256, 256, 0, stream>>>(a_l1_w[l], w1h[l], 64 * 128);
    }
    k_inputnet<<<(n * HDIM + 255) / 256, 256, 0, stream>>>(x, datanorm, in_w, in_b, hA, hAh, n);

    // ---- two aggregation layers ----
    int* vcur = validA;
    int* vnext = validB;
    for (int l = 0; l < 2; ++l) {
        k_sqnorm<<<(n + 255) / 256, 256, 0, stream>>>(hA, sq, n);
        k_knn<<<n / 16, 32, 0, stream>>>(hAh, sq, batch, vcur, gstart, idx);
        k_edgeconv<<<n, 32, 0, stream>>>(hA, idx, vcur,
                                         w0h[l], a_l0_b[l], a_bn0_g[l], a_bn0_b[l],
                                         w1h[l], a_l1_b[l], a_bn1_g[l], a_bn1_b[l],
                                         hB);
        k_partner<<<(n + 255) / 256, 256, 0, stream>>>(hB, idx, vcur, partner, n);
        k_pool<<<(n * HDIM + 255) / 256, 256, 0, stream>>>(hB, partner, vcur, hA, hAh, vnext, n);
        int* tmp = vcur; vcur = vnext; vnext = tmp;
    }

    // ---- head ----
    k_output<<<1, NGRAPH * HDIM, 0, stream>>>(hA, vcur, gstart,
                                              o0_w, o0_b, o1_w, o1_b, o2_w, o2_b, out);
}
